// MultiHeadAttention_11484742550087
// MI455X (gfx1250) — matrix operand order repaired and checked by reading, not compiled, not run on hardware
//
#include <hip/hip_runtime.h>
#include <stdint.h>

typedef __attribute__((ext_vector_type(16))) _Float16 v16h;
typedef __attribute__((ext_vector_type(8)))  _Float16 h8;
typedef __attribute__((ext_vector_type(8)))  float    v8f;

#define BATCH   2
#define SEQ     2048
#define DMODEL  1024
#define NHEADS  16
#define HEADDIM 64
#define NEGVAL  (-1e10f)

#define WMMA_F16(a, b, c) \
    __builtin_amdgcn_wmma_f32_16x16x32_f16(false, (a), false, (b), (short)0, (c), false, false)

// ---------------------------------------------------------------------------
// Fragment helpers for v_wmma_f32_16x16x32_f16 (ISA 7.12.2):
//  A 16x32: lane<16 -> M=lane, K=0..15; lane>=16 -> K=16..31 (contiguous k, same order as B)
//  B 32x16: lane<16 -> N=lane, K=0..15;         lane>=16 -> K=16..31
//  C/D:     vgpr r -> M = 8*(lane>=16)+r, N = lane&15
// ---------------------------------------------------------------------------
__device__ __forceinline__ v16h frag_combine(h8 lo, h8 hi) {
    v16h a;
#pragma unroll
    for (int i = 0; i < 8; ++i) { a[i] = lo[i]; a[8 + i] = hi[i]; }
    return a;
}

// A fragment from row-major f16 global (K contiguous)
__device__ __forceinline__ v16h frag_a_f16g(const _Float16* A, int row0, int k0, int ld) {
    int lane = threadIdx.x & 31;
    int m  = lane & 15;
    int kb = (lane >> 4) << 4;
    const _Float16* p = A + (size_t)(row0 + m) * ld + k0 + kb;
    return frag_combine(*(const h8*)p, *(const h8*)(p + 8));
}

// B fragment where K runs along the contiguous axis of row-major [n][k] f16
// (used for Q*K^T: K-matrix rows are the B columns)
__device__ __forceinline__ v16h frag_b_kcontig(const _Float16* Bm, int k0, int col0, int ld) {
    int lane = threadIdx.x & 31;
    int n  = lane & 15;
    int kb = (lane >> 4) << 4;
    const _Float16* p = Bm + (size_t)(col0 + n) * ld + k0 + kb;
    return frag_combine(*(const h8*)p, *(const h8*)(p + 8));
}

// ---------------------------------------------------------------------------
// Async global->LDS helpers (CDNA5 §08, GLOBAL_LOAD_ASYNC_TO_LDS_B128, GV mode)
// Copies a 32x64-half (4 KB) V slab: 8 issues x 32 lanes x 16B.
// ---------------------------------------------------------------------------
__device__ __forceinline__ void async_stage_v32x64(const _Float16* vbase, unsigned ldsoff,
                                                   int lane) {
    const int r   = lane >> 3;   // 0..3
    const int seg = lane & 7;    // 16B segment within a 128B row
#pragma unroll
    for (int j = 0; j < 8; ++j) {
        const int row = j * 4 + r;
        const char* g = (const char*)vbase + (size_t)row * (DMODEL * 2) + seg * 16;
        unsigned l = ldsoff + row * 128 + seg * 16;
        asm volatile("global_load_async_to_lds_b128 %0, %1, off"
                     :: "v"(l), "v"(g) : "memory");
    }
}
__device__ __forceinline__ void wait_async_le8() {
    asm volatile("s_wait_asynccnt 0x8" ::: "memory");
}
__device__ __forceinline__ void wait_async_0() {
    asm volatile("s_wait_asynccnt 0x0" ::: "memory");
}

// ---------------------------------------------------------------------------
// Block-tiled GEMM: out[M,N] = X[M,K] @ W_f32[K,N] + bias  (optionally *scale,
// f16 output). 64x64 block, 256 threads (8 waves), 2 accum tiles per wave.
// A/W tiles staged to LDS as f16 (convert once per element), fragments are
// aligned ds_load_b128s.
// ---------------------------------------------------------------------------
template <typename AT, bool OUT_F16>
__global__ __launch_bounds__(256) void gemm_kernel(
    const AT* __restrict__ X, const float* __restrict__ W,
    const float* __restrict__ bias, void* __restrict__ outp, float scale) {
    constexpr int N = DMODEL, K = DMODEL;
    __shared__ __align__(16) _Float16 As[64][32];  // [m][k]   4 KB
    __shared__ __align__(16) _Float16 Bs[64][40];  // [n][k]+pad 5 KB

    const int mt = blockIdx.x >> 4;   // N/64 = 16 tiles
    const int nt = blockIdx.x & 15;
    const int row0 = mt * 64, col0 = nt * 64;
    const int tid = threadIdx.x, lane = tid & 31, wv = tid >> 5;
    const int m16 = (wv & 3) << 4;    // wave's M sub-tile
    const int nw  = (wv >> 2) << 5;   // wave's 32-wide N range

    const int sm = tid >> 2, skq = (tid & 3) << 3;    // A staging coords
    const int sn = tid & 63, skk0 = (tid >> 6) << 3;  // B staging coords

    v8f c0 = {}, c1 = {};

    for (int k0 = 0; k0 < K; k0 += 32) {
        __syncthreads();
        {   // stage A tile: 8 contiguous elems per thread, one b128 LDS store
            const AT* p = X + (size_t)(row0 + sm) * K + k0 + skq;
            h8 v;
#pragma unroll
            for (int i = 0; i < 8; ++i) v[i] = (_Float16)p[i];
            *(h8*)&As[sm][skq] = v;
        }
        {   // stage B tile transposed [n][k]; reads coalesce along n
#pragma unroll
            for (int i = 0; i < 8; ++i)
                Bs[sn][skk0 + i] = (_Float16)W[(size_t)(k0 + skk0 + i) * N + col0 + sn];
            if (k0 + 32 < K)
                __builtin_prefetch(&W[(size_t)(k0 + 32 + skk0) * N + col0 + sn], 0, 0);
        }
        __syncthreads();

        const int ml = m16 + (lane & 15);
        const int kb = (lane >> 4) << 4;
        v16h a = frag_combine(*(const h8*)&As[ml][kb], *(const h8*)&As[ml][kb + 8]);
        const int nn  = lane & 15;
        const int kb2 = (lane >> 4) << 4;
        v16h b0 = frag_combine(*(const h8*)&Bs[nw + nn][kb2],
                               *(const h8*)&Bs[nw + nn][kb2 + 8]);
        v16h b1 = frag_combine(*(const h8*)&Bs[nw + 16 + nn][kb2],
                               *(const h8*)&Bs[nw + 16 + nn][kb2 + 8]);
        c0 = WMMA_F16(a, b0, c0);
        c1 = WMMA_F16(a, b1, c1);
    }

    const int nn = lane & 15, mb = (lane >> 4) << 3;
#pragma unroll
    for (int half = 0; half < 2; ++half) {
        v8f& c = half ? c1 : c0;
        const int col = col0 + nw + half * 16 + nn;
        const float bv = bias[col];
#pragma unroll
        for (int r = 0; r < 8; ++r) {
            const size_t idx = (size_t)(row0 + m16 + mb + r) * N + col;
            const float v = c[r] + bv;
            if (OUT_F16) ((_Float16*)outp)[idx] = (_Float16)(v * scale);
            else         ((float*)outp)[idx]    = v;
        }
    }
}

// ---------------------------------------------------------------------------
// Fused attention: one block (8 waves) per (b, h, 16-row q-tile).
//  phase 1: scores (QK^T, 1/sqrt(D) folded into Q) + mask -> LDS [16][2048]
//  phase 2: row softmax; normalized P -> softmax_out and back to LDS
//  phase 3: P @ V; per-wave 256-key partials with double-buffered async
//           global->LDS V staging; LDS reduce -> ctx (f16)
// ---------------------------------------------------------------------------
__global__ __launch_bounds__(256) void attn_kernel(
    const _Float16* __restrict__ qh, const _Float16* __restrict__ kh,
    const _Float16* __restrict__ vh, const int* __restrict__ mask,
    float* __restrict__ softmax_out, _Float16* __restrict__ ctx) {

    __shared__ float sc[16][SEQ];                                   // 128 KB
    __shared__ float part[8][16][HEADDIM];                          //  32 KB
    __shared__ __align__(16) _Float16 vstage[8][2][32][HEADDIM];    //  64 KB
    __shared__ float red[2][16][16];
    __shared__ float rowstat[2][16];

    const int qt = blockIdx.x & (SEQ / 16 - 1);
    const int bh = blockIdx.x >> 7;
    const int h  = bh & (NHEADS - 1);
    const int b  = bh >> 4;
    const int q0 = qt * 16;

    const int tid  = threadIdx.x;
    const int lane = tid & 31;
    const int wv   = tid >> 5;       // 0..7
    const int hoff = h * HEADDIM;
    const int kc0  = wv * 256;       // this wave's key chunk

    // ---- phase 1: scores -------------------------------------------------
    const v16h aq0 = frag_a_f16g(qh + hoff, b * SEQ + q0, 0,  DMODEL);
    const v16h aq1 = frag_a_f16g(qh + hoff, b * SEQ + q0, 32, DMODEL);
    for (int t = 0; t < 16; ++t) {
        const int key0 = kc0 + t * 16;
        v8f c = {};
        v16h b0 = frag_b_kcontig(kh + hoff, 0,  b * SEQ + key0, DMODEL);
        v16h b1 = frag_b_kcontig(kh + hoff, 32, b * SEQ + key0, DMODEL);
        c = WMMA_F16(aq0, b0, c);
        c = WMMA_F16(aq1, b1, c);
        const int n = lane & 15, mb = (lane >> 4) << 3;
#pragma unroll
        for (int r = 0; r < 8; ++r) {
            const int qrow = q0 + mb + r;
            const int kcol = key0 + n;
            const int mv = mask[((size_t)b * SEQ + qrow) * SEQ + kcol];
            sc[mb + r][kcol] = (mv == 0) ? NEGVAL : c[r];
        }
    }
    __syncthreads();

    // ---- phase 2: softmax ------------------------------------------------
    const int row = tid >> 4, sub = tid & 15;
    float pm = -3.4e38f;
    for (int k = sub; k < SEQ; k += 16) pm = fmaxf(pm, sc[row][k]);
    red[0][row][sub] = pm;
    __syncthreads();
    if (sub == 0) {
        float mm = -3.4e38f;
#pragma unroll
        for (int i = 0; i < 16; ++i) mm = fmaxf(mm, red[0][row][i]);
        rowstat[0][row] = mm;
    }
    __syncthreads();
    const float rm = rowstat[0][row];
    float ps = 0.f;
    for (int k = sub; k < SEQ; k += 16) {
        float e = __expf(sc[row][k] - rm);
        sc[row][k] = e;
        ps += e;
    }
    red[1][row][sub] = ps;
    __syncthreads();
    if (sub == 0) {
        float ss = 0.f;
#pragma unroll
        for (int i = 0; i < 16; ++i) ss += red[1][row][i];
        rowstat[1][row] = ss;
    }
    __syncthreads();
    const float inv = 1.f / rowstat[1][row];
    float* srow = softmax_out + (((size_t)(b * NHEADS + h) * SEQ) + q0 + row) * SEQ;
    for (int k = sub; k < SEQ; k += 16) {
        float p = sc[row][k] * inv;
        sc[row][k] = p;
        srow[k] = p;
    }
    __syncthreads();

    // ---- phase 3: P @ V with double-buffered async V staging -------------
    const _Float16* vchunk = vh + (size_t)(b * SEQ + kc0) * DMODEL + hoff;
    const unsigned vso[2] = { (unsigned)(uintptr_t)&vstage[wv][0][0][0],
                              (unsigned)(uintptr_t)&vstage[wv][1][0][0] };
    async_stage_v32x64(vchunk, vso[0], lane);

    v8f acc[4] = {{}, {}, {}, {}};
    for (int kt = 0; kt < 8; ++kt) {
        const int k0 = kc0 + kt * 32;
        if (kt < 7) {
            async_stage_v32x64(vchunk + (size_t)(kt + 1) * 32 * DMODEL,
                               vso[(kt + 1) & 1], lane);
            wait_async_le8();   // previous slab complete (in-order async)
        } else {
            wait_async_0();
        }
        // A fragment from LDS probabilities (f32 -> f16)
        const int m = lane & 15, kb = (lane >> 4) << 4;
        v16h a;
#pragma unroll
        for (int i = 0; i < 8; ++i) {
            a[i]     = (_Float16)sc[m][k0 + kb + i];
            a[8 + i] = (_Float16)sc[m][k0 + kb + 8 + i];
        }
        const int nn = lane & 15, kb2 = (lane >> 4) << 4;
        const _Float16 (*vs)[HEADDIM] = vstage[wv][kt & 1];
#pragma unroll
        for (int ntile = 0; ntile < 4; ++ntile) {
            v16h bb;
#pragma unroll
            for (int i = 0; i < 16; ++i) bb[i] = vs[kb2 + i][ntile * 16 + nn];
            acc[ntile] = WMMA_F16(a, bb, acc[ntile]);
        }
    }
    {
        const int nn = lane & 15, mb = (lane >> 4) << 3;
#pragma unroll
        for (int ntile = 0; ntile < 4; ++ntile)
#pragma unroll
            for (int r = 0; r < 8; ++r)
                part[wv][mb + r][ntile * 16 + nn] = acc[ntile][r];
    }
    __syncthreads();

    for (int idx = tid; idx < 16 * HEADDIM; idx += 256) {
        const int m = idx >> 6, d = idx & (HEADDIM - 1);
        float s = 0.f;
#pragma unroll
        for (int w = 0; w < 8; ++w) s += part[w][m][d];
        ctx[((size_t)(b * SEQ + q0 + m)) * DMODEL + hoff + d] = (_Float16)s;
    }
}

// ---------------------------------------------------------------------------
extern "C" void kernel_launch(void* const* d_in, const int* in_sizes, int n_in,
                              void* d_out, int out_size, void* d_ws, size_t ws_size,
                              hipStream_t stream) {
    const float* query = (const float*)d_in[0];
    const float* key   = (const float*)d_in[1];
    const float* value = (const float*)d_in[2];
    const int*   mask  = (const int*)d_in[3];
    const float* Wq = (const float*)d_in[4];
    const float* bq = (const float*)d_in[5];
    const float* Wk = (const float*)d_in[6];
    const float* bk = (const float*)d_in[7];
    const float* Wv = (const float*)d_in[8];
    const float* bv = (const float*)d_in[9];
    const float* Wo = (const float*)d_in[10];
    const float* bo = (const float*)d_in[11];

    float* out         = (float*)d_out;                       // [B,S,D]
    float* softmax_out = out + (size_t)BATCH * SEQ * DMODEL;  // [B,H,S,S]

    const size_t BSD = (size_t)BATCH * SEQ * DMODEL;
    _Float16* qh  = (_Float16*)d_ws;
    _Float16* kh  = qh + BSD;
    _Float16* vh  = kh + BSD;
    _Float16* ctx = vh + BSD;

    const int M = BATCH * SEQ;                        // 4096
    dim3 gemmGrid((M / 64) * (DMODEL / 64));          // 1024 blocks x 256 thr

    const float qscale = 0.03125f;                    // 1/sqrt(D_MODEL)
    gemm_kernel<float, true><<<gemmGrid, 256, 0, stream>>>(query, Wq, bq, qh, qscale);
    gemm_kernel<float, true><<<gemmGrid, 256, 0, stream>>>(key,   Wk, bk, kh, 1.0f);
    gemm_kernel<float, true><<<gemmGrid, 256, 0, stream>>>(value, Wv, bv, vh, 1.0f);

    attn_kernel<<<dim3(BATCH * NHEADS * (SEQ / 16)), 256, 0, stream>>>(
        qh, kh, vh, mask, softmax_out, ctx);

    gemm_kernel<_Float16, false><<<gemmGrid, 256, 0, stream>>>(ctx, Wo, bo, out, 1.0f);
}
